// Encoder_12060268167528
// MI455X (gfx1250) — matrix operand order repaired and checked by reading, compiled, not run on hardware
//
#include <hip/hip_runtime.h>
#include <math.h>

// ---------------------------------------------------------------------------
// Types for CDNA5 WMMA (wave32): v_wmma_f32_16x16x32_bf16
// ---------------------------------------------------------------------------
typedef __attribute__((ext_vector_type(16))) __bf16 v16bf;
typedef __attribute__((ext_vector_type(8)))  __bf16 v8bf;
typedef __attribute__((ext_vector_type(8)))  float  v8f;

__device__ __forceinline__ v8f wmma_bf16(v16bf a, v16bf b, v8f c) {
    // 8 args: (neg_a, A, neg_b, B, c_mod, C, reuse_a, reuse_b)
    return __builtin_amdgcn_wmma_f32_16x16x32_bf16(false, a, false, b,
                                                   (short)0, c, false, false);
}

// A-matrix 16x32 bf16 fragment. Lane layout used here (contiguous k, same order as load_b_frag):
//   row m = lane%16, half = lane/16
//   elements 0..7  -> K = half*16 + e       (contiguous 16B)
//   elements 8..15 -> K = half*16 + e       (contiguous 16B)
// `base` points at tile element (m=0, k=0); row stride = ld (elements).
__device__ __forceinline__ v16bf load_a_frag(const __bf16* base, int ld, int lane) {
    int m  = lane & 15;
    int hl = lane >> 4;
    const __bf16* p = base + (size_t)m * ld + hl * 16;
    v8bf lo = *(const v8bf*)(p);
    v8bf hi = *(const v8bf*)(p + 8);
    return __builtin_shufflevector(lo, hi, 0,1,2,3,4,5,6,7,8,9,10,11,12,13,14,15);
}

__device__ __forceinline__ void prefetch_a(const __bf16* base, int ld, int lane) {
    int m  = lane & 15;
    int hl = lane >> 4;
    __builtin_prefetch(base + (size_t)m * ld + hl * 8, 0, 1); // global_prefetch_b8
}

// B-matrix 32x16 bf16 fragment from a [N,K] (pre-transposed) buffer.
// Lane layout: col n = lane%16, half = lane/16, elements e -> K = half*16 + e
// -> one contiguous 32B chunk per lane.
__device__ __forceinline__ v16bf load_b_frag(const __bf16* baseT, int ld, int lane) {
    int n  = lane & 15;
    int hl = lane >> 4;
    return *(const v16bf*)(baseT + (size_t)n * ld + hl * 16);
}

__device__ __forceinline__ void prefetch_b(const __bf16* baseT, int ld, int lane) {
    int n  = lane & 15;
    int hl = lane >> 4;
    __builtin_prefetch(baseT + (size_t)n * ld + hl * 16, 0, 1);
}

// ---------------------------------------------------------------------------
// Embedding: h[row,:] = tok_emb[x[row],:] + pos_emb[row%1024,:]   (fp32)
// ---------------------------------------------------------------------------
__global__ __launch_bounds__(256)
void embed_kernel(const int* __restrict__ x, const float* __restrict__ tok,
                  const float* __restrict__ pos, float* __restrict__ h) {
    int row = blockIdx.x;               // 0..8191  (= b*1024 + t)
    int tid = threadIdx.x;
    int token = x[row];
    int t = row & 1023;
    float*       hr = h   + (size_t)row   * 768;
    const float* te = tok + (size_t)token * 768;
    const float* pe = pos + (size_t)t     * 768;
    hr[tid]       = te[tid]       + pe[tid];
    hr[tid + 256] = te[tid + 256] + pe[tid + 256];
    hr[tid + 512] = te[tid + 512] + pe[tid + 512];
}

// ---------------------------------------------------------------------------
// LayerNorm over D=768, one token per block, bf16 output (GEMM A input)
// ---------------------------------------------------------------------------
__global__ __launch_bounds__(256)
void ln_kernel(const float* __restrict__ x, const float* __restrict__ g,
               const float* __restrict__ b, __bf16* __restrict__ out) {
    __shared__ float red[256];
    int row = blockIdx.x;
    int tid = threadIdx.x;
    const float* xr = x + (size_t)row * 768;
    float v0 = xr[tid], v1 = xr[tid + 256], v2 = xr[tid + 512];

    red[tid] = v0 + v1 + v2;
    __syncthreads();
    for (int off = 128; off > 0; off >>= 1) {
        if (tid < off) red[tid] += red[tid + off];
        __syncthreads();
    }
    float mu = red[0] * (1.0f / 768.0f);
    __syncthreads();

    float d0 = v0 - mu, d1 = v1 - mu, d2 = v2 - mu;
    red[tid] = d0 * d0 + d1 * d1 + d2 * d2;
    __syncthreads();
    for (int off = 128; off > 0; off >>= 1) {
        if (tid < off) red[tid] += red[tid + off];
        __syncthreads();
    }
    float rstd = rsqrtf(red[0] * (1.0f / 768.0f) + 1e-5f);

    __bf16* orow = out + (size_t)row * 768;
    orow[tid]       = (__bf16)(d0 * rstd * g[tid]       + b[tid]);
    orow[tid + 256] = (__bf16)(d1 * rstd * g[tid + 256] + b[tid + 256]);
    orow[tid + 512] = (__bf16)(d2 * rstd * g[tid + 512] + b[tid + 512]);
}

// ---------------------------------------------------------------------------
// Weight fp32 [K,N]  ->  bf16 transposed [N,K]   (tiled via LDS)
// ---------------------------------------------------------------------------
__global__ __launch_bounds__(256)
void transpose_convert(const float* __restrict__ in, __bf16* __restrict__ out,
                       int K, int N) {
    __shared__ float tile[32][33];
    int n0 = blockIdx.x * 32;
    int k0 = blockIdx.y * 32;
    int tx = threadIdx.x & 31;
    int ty = threadIdx.x >> 5;                    // 0..7
#pragma unroll
    for (int r = 0; r < 4; ++r)
        tile[ty + r * 8][tx] = in[(size_t)(k0 + ty + r * 8) * N + n0 + tx];
    __syncthreads();
#pragma unroll
    for (int r = 0; r < 4; ++r)
        out[(size_t)(n0 + ty + r * 8) * K + k0 + tx] = (__bf16)tile[tx][ty + r * 8];
}

__global__ __launch_bounds__(256)
void convert_bf16(const float* __restrict__ in, __bf16* __restrict__ out, int n) {
    int i = blockIdx.x * 256 + threadIdx.x;
    if (i < n) out[i] = (__bf16)in[i];
}

// ---------------------------------------------------------------------------
// WMMA GEMM:  C[M,N] = A[M,K] (bf16) x Bt[N,K] (bf16, pre-transposed) + epi
//   block = 256 threads = 8 waves, block tile = 256(M) x 64(N)
//   wave tile = 32(M) x 64(N) = 2x4 WMMA tiles, 8 WMMAs per K-step
//   All 8 waves share the same B-tiles -> WGP$ reuse; prefetch 8 K-steps ahead.
// Epilogues:
//   0: out fp32 = acc (+bias if non-null)                       [logits]
//   1: out fp32 = acc + bias + residual (in place on h)         [proj/ff2]
//   2: out bf16 = gelu_exact(acc + bias)                        [ff1]
//   3: scatter acc+bias -> q[b,h,t,hd], k[b,h,t,hd], vT[b,h,hd,t] (bf16)
// ---------------------------------------------------------------------------
template <int EPI>
__global__ __launch_bounds__(256)
void gemm_kernel(const __bf16* __restrict__ A, const __bf16* __restrict__ Bt,
                 const float* __restrict__ bias, const float* __restrict__ res,
                 float* __restrict__ outf, __bf16* __restrict__ outb,
                 __bf16* __restrict__ qb, __bf16* __restrict__ kb,
                 __bf16* __restrict__ vtb, int M, int N, int K) {
    int lane = threadIdx.x & 31;
    int wave = threadIdx.x >> 5;
    int rbase = blockIdx.y * 256 + wave * 32;
    int cbase = blockIdx.x * 64;

    const int PF = 256;   // prefetch distance in K elements (8 K-steps, 512B)

    v8f acc[2][4] = {};

    for (int kk = 0; kk < K; kk += 32) {
        v16bf a0 = load_a_frag(A + (size_t)rbase * K + kk,        K, lane);
        v16bf a1 = load_a_frag(A + (size_t)(rbase + 16) * K + kk, K, lane);
        v16bf bfrag[4];
#pragma unroll
        for (int nt = 0; nt < 4; ++nt)
            bfrag[nt] = load_b_frag(Bt + (size_t)(cbase + nt * 16) * K + kk, K, lane);

        if (kk + PF < K) {
            prefetch_a(A + (size_t)rbase * K + kk + PF,        K, lane);
            prefetch_a(A + (size_t)(rbase + 16) * K + kk + PF, K, lane);
#pragma unroll
            for (int nt = 0; nt < 4; ++nt)
                prefetch_b(Bt + (size_t)(cbase + nt * 16) * K + kk + PF, K, lane);
        }

#pragma unroll
        for (int nt = 0; nt < 4; ++nt) {
            acc[0][nt] = wmma_bf16(a0, bfrag[nt], acc[0][nt]);
            acc[1][nt] = wmma_bf16(a1, bfrag[nt], acc[1][nt]);
        }
    }

    // C layout: VGPR r holds row m = r + 8*(lane/16), col = lane%16
    int hl = lane >> 4;
    int nloc = lane & 15;
#pragma unroll
    for (int mt = 0; mt < 2; ++mt) {
#pragma unroll
        for (int nt = 0; nt < 4; ++nt) {
#pragma unroll
            for (int r = 0; r < 8; ++r) {
                int row = rbase + mt * 16 + r + 8 * hl;
                int col = cbase + nt * 16 + nloc;
                float v = acc[mt][nt][r];
                if (bias) v += bias[col];
                if (EPI == 0) {
                    outf[(size_t)row * N + col] = v;
                } else if (EPI == 1) {
                    outf[(size_t)row * N + col] = v + res[(size_t)row * N + col];
                } else if (EPI == 2) {
                    float ge = 0.5f * v * (1.0f + erff(v * 0.70710678118654752f));
                    outb[(size_t)row * N + col] = (__bf16)ge;
                } else { // EPI == 3 : QKV scatter (N == 2304)
                    int b_ = row >> 10;
                    int t  = row & 1023;
                    if (col < 768) {
                        int h_ = col >> 6, hd = col & 63;
                        qb[(((size_t)(b_ * 12 + h_) * 1024) + t) * 64 + hd] = (__bf16)v;
                    } else if (col < 1536) {
                        int c2 = col - 768;
                        int h_ = c2 >> 6, hd = c2 & 63;
                        kb[(((size_t)(b_ * 12 + h_) * 1024) + t) * 64 + hd] = (__bf16)v;
                    } else {
                        int c2 = col - 1536;
                        int h_ = c2 >> 6, hd = c2 & 63;
                        vtb[((size_t)(b_ * 12 + h_) * 64 + hd) * 1024 + t] = (__bf16)v;
                    }
                }
            }
        }
    }
}

// ---------------------------------------------------------------------------
// Flash attention via transposed tiles (all-WMMA):
//   S^T = K * Q^T    (A = K rows, B = Q^T; softmax axis lands per-lane)
//   O^T = V^T * P^T  (A = V^T rows from pre-transposed V, B = exp'd S^T)
// One wave = 16 queries; block = 8 waves = 128 queries; grid = (8, B*H)
// All 8 waves stream identical K/V chunks -> WGP$ reuse + 1-chunk-ahead
// prefetch to overlap with softmax TRANS work.
// ---------------------------------------------------------------------------
__global__ __launch_bounds__(256)
void attn_kernel(const __bf16* __restrict__ Q, const __bf16* __restrict__ Kbuf,
                 const __bf16* __restrict__ Vt, __bf16* __restrict__ O) {
    int lane = threadIdx.x & 31;
    int wave = threadIdx.x >> 5;
    int bh = blockIdx.y;                 // 0..95
    int qbase = blockIdx.x * 128 + wave * 16;
    int n  = lane & 15;                  // query within wave tile
    int hl = lane >> 4;

    const __bf16* qp    = Q    + ((size_t)bh * 1024 + qbase) * 64;
    const __bf16* kbase = Kbuf + (size_t)bh * 1024 * 64;
    const __bf16* vbase = Vt   + (size_t)bh * 64 * 1024;

    // Q^T B-fragments: col n = query, elements contiguous in hd
    v16bf bq0 = *(const v16bf*)(qp + (size_t)n * 64 + hl * 16);
    v16bf bq1 = *(const v16bf*)(qp + (size_t)n * 64 + 32 + hl * 16);

    v8f o0 = {}, o1 = {}, o2 = {}, o3 = {};
    float m_i = -3.0e38f, l_i = 0.0f;

    for (int kc = 0; kc < 1024; kc += 32) {
        // --- S^T tiles: keys [kc..kc+15] and [kc+16..kc+31], over hd=64 ---
        v8f c0 = {}, c1 = {};
        {
            v16bf ka0 = load_a_frag(kbase + (size_t)kc * 64,             64, lane);
            v16bf ka1 = load_a_frag(kbase + (size_t)kc * 64 + 32,        64, lane);
            c0 = wmma_bf16(ka0, bq0, c0);
            c0 = wmma_bf16(ka1, bq1, c0);
            v16bf kb0 = load_a_frag(kbase + (size_t)(kc + 16) * 64,      64, lane);
            v16bf kb1 = load_a_frag(kbase + (size_t)(kc + 16) * 64 + 32, 64, lane);
            c1 = wmma_bf16(kb0, bq0, c1);
            c1 = wmma_bf16(kb1, bq1, c1);
        }

        // prefetch next chunk's K rows and V^T rows while softmax runs
        if (kc + 32 < 1024) {
            prefetch_a(kbase + (size_t)(kc + 32) * 64,      64, lane);
            prefetch_a(kbase + (size_t)(kc + 48) * 64,      64, lane);
            prefetch_a(vbase + (size_t)0  * 1024 + kc + 32, 1024, lane);
            prefetch_a(vbase + (size_t)16 * 1024 + kc + 32, 1024, lane);
            prefetch_a(vbase + (size_t)32 * 1024 + kc + 32, 1024, lane);
            prefetch_a(vbase + (size_t)48 * 1024 + kc + 32, 1024, lane);
        }

        // --- online softmax (per query = per lane + one xor-16 combine) ---
        float p0[8], p1[8];
        float cm = -3.0e38f;
#pragma unroll
        for (int r = 0; r < 8; ++r) {
            p0[r] = c0[r] * 0.125f;      // 1/sqrt(64)
            p1[r] = c1[r] * 0.125f;
            cm = fmaxf(cm, fmaxf(p0[r], p1[r]));
        }
        cm = fmaxf(cm, __shfl_xor(cm, 16, 32));
        float mnew = fmaxf(m_i, cm);
        float corr = __expf(m_i - mnew);
        float ls = 0.0f;
#pragma unroll
        for (int r = 0; r < 8; ++r) {
            p0[r] = __expf(p0[r] - mnew);
            p1[r] = __expf(p1[r] - mnew);
            ls += p0[r] + p1[r];
        }
        ls += __shfl_xor(ls, 16, 32);
        l_i = l_i * corr + ls;
        m_i = mnew;
#pragma unroll
        for (int r = 0; r < 8; ++r) {
            o0[r] *= corr; o1[r] *= corr; o2[r] *= corr; o3[r] *= corr;
        }

        // --- build P^T B-fragment (32 keys x 16 queries) ---
        // half0 lane needs keys 0..15:  own p0[0..7] + partner(c0)[0..7]
        // half1 lane needs keys 16..31: partner(c1)[0..7] + own p1[0..7]
        v16bf pb;
#pragma unroll
        for (int r = 0; r < 8; ++r) {
            float q0v = __shfl_xor(p0[r], 16, 32);
            float q1v = __shfl_xor(p1[r], 16, 32);
            float lo = hl ? q1v   : p0[r];
            float hi = hl ? p1[r] : q0v;
            pb[r]     = (__bf16)lo;
            pb[8 + r] = (__bf16)hi;
        }

        // --- O^T += V^T * P^T, four hd-chunks of 16 ---
        v16bf va;
        va = load_a_frag(vbase + (size_t)0  * 1024 + kc, 1024, lane); o0 = wmma_bf16(va, pb, o0);
        va = load_a_frag(vbase + (size_t)16 * 1024 + kc, 1024, lane); o1 = wmma_bf16(va, pb, o1);
        va = load_a_frag(vbase + (size_t)32 * 1024 + kc, 1024, lane); o2 = wmma_bf16(va, pb, o2);
        va = load_a_frag(vbase + (size_t)48 * 1024 + kc, 1024, lane); o3 = wmma_bf16(va, pb, o3);
    }

    // --- write O (bf16) to [B,T,D] layout for the proj GEMM ---
    float inv = 1.0f / l_i;
    int b_ = bh / 12, h_ = bh % 12;
    int t = qbase + n;
    __bf16* op = O + ((size_t)b_ * 1024 + t) * 768 + h_ * 64;
#pragma unroll
    for (int r = 0; r < 8; ++r) {
        op[ 0 + r + 8 * hl] = (__bf16)(o0[r] * inv);
        op[16 + r + 8 * hl] = (__bf16)(o1[r] * inv);
        op[32 + r + 8 * hl] = (__bf16)(o2[r] * inv);
        op[48 + r + 8 * hl] = (__bf16)(o3[r] * inv);
    }
}

// ---------------------------------------------------------------------------
// Host-side orchestration
// ---------------------------------------------------------------------------
extern "C" void kernel_launch(void* const* d_in, const int* in_sizes, int n_in,
                              void* d_out, int out_size, void* d_ws, size_t ws_size,
                              hipStream_t stream) {
    (void)in_sizes; (void)n_in; (void)out_size; (void)ws_size;

    const int*   x     = (const int*)  d_in[0];
    const float* tok   = (const float*)d_in[1];
    const float* pos   = (const float*)d_in[2];
    const float* ln1g  = (const float*)d_in[3];
    const float* ln1b  = (const float*)d_in[4];
    const float* Wqkv  = (const float*)d_in[5];
    const float* bqkv  = (const float*)d_in[6];
    const float* Wproj = (const float*)d_in[7];
    const float* bproj = (const float*)d_in[8];
    const float* ln2g  = (const float*)d_in[9];
    const float* ln2b  = (const float*)d_in[10];
    const float* Wff1  = (const float*)d_in[11];
    const float* bff1  = (const float*)d_in[12];
    const float* Wff2  = (const float*)d_in[13];
    const float* bff2  = (const float*)d_in[14];
    const float* lnfg  = (const float*)d_in[15];
    const float* lnfb  = (const float*)d_in[16];

    const int M = 8192;          // B*T
    char* ws = (char*)d_ws;
    size_t off = 0;
    auto alloc = [&](size_t bytes) -> void* {
        void* p = ws + off;
        off = (off + bytes + 255) & ~(size_t)255;
        return p;
    };

    float*  h     = (float*) alloc((size_t)M * 768 * 4);
    __bf16* hn    = (__bf16*)alloc((size_t)M * 768 * 2);
    __bf16* qb    = (__bf16*)alloc((size_t)M * 768 * 2);
    __bf16* kb    = (__bf16*)alloc((size_t)M * 768 * 2);
    __bf16* vtb   = (__bf16*)alloc((size_t)M * 768 * 2);
    __bf16* ob    = (__bf16*)alloc((size_t)M * 768 * 2);
    __bf16* ffb   = (__bf16*)alloc((size_t)M * 3072 * 2);
    __bf16* wqkvT = (__bf16*)alloc((size_t)768 * 2304 * 2);
    __bf16* wprjT = (__bf16*)alloc((size_t)768 * 768 * 2);
    __bf16* wff1T = (__bf16*)alloc((size_t)768 * 3072 * 2);
    __bf16* wff2T = (__bf16*)alloc((size_t)3072 * 768 * 2);
    __bf16* tokb  = (__bf16*)alloc((size_t)256 * 768 * 2);

    embed_kernel<<<M, 256, 0, stream>>>(x, tok, pos, h);

    for (int l = 0; l < 12; ++l) {
        transpose_convert<<<dim3(2304/32,  768/32), 256, 0, stream>>>(
            Wqkv  + (size_t)l * 768 * 2304, wqkvT, 768, 2304);
        transpose_convert<<<dim3( 768/32,  768/32), 256, 0, stream>>>(
            Wproj + (size_t)l * 768 * 768,  wprjT, 768, 768);
        transpose_convert<<<dim3(3072/32,  768/32), 256, 0, stream>>>(
            Wff1  + (size_t)l * 768 * 3072, wff1T, 768, 3072);
        transpose_convert<<<dim3( 768/32, 3072/32), 256, 0, stream>>>(
            Wff2  + (size_t)l * 3072 * 768, wff2T, 3072, 768);

        // hn = LN1(h)
        ln_kernel<<<M, 256, 0, stream>>>(h, ln1g + l * 768, ln1b + l * 768, hn);

        // qkv = hn @ Wqkv + bqkv  -> q/k (b,h,t,hd), vT (b,h,hd,t)
        gemm_kernel<3><<<dim3(2304/64, M/256), 256, 0, stream>>>(
            hn, wqkvT, bqkv + (size_t)l * 2304, nullptr, nullptr, nullptr,
            qb, kb, vtb, M, 2304, 768);

        // attention -> ob (bf16 [B,T,D])
        attn_kernel<<<dim3(8, 96), 256, 0, stream>>>(qb, kb, vtb, ob);

        // h = h + ob @ Wproj + bproj   (in place on h)
        gemm_kernel<1><<<dim3(768/64, M/256), 256, 0, stream>>>(
            ob, wprjT, bproj + (size_t)l * 768, h, h, nullptr,
            nullptr, nullptr, nullptr, M, 768, 768);

        // hn = LN2(h)
        ln_kernel<<<M, 256, 0, stream>>>(h, ln2g + l * 768, ln2b + l * 768, hn);

        // ffb = gelu(hn @ Wff1 + bff1)   (bf16)
        gemm_kernel<2><<<dim3(3072/64, M/256), 256, 0, stream>>>(
            hn, wff1T, bff1 + (size_t)l * 3072, nullptr, nullptr, ffb,
            nullptr, nullptr, nullptr, M, 3072, 768);

        // h = h + ffb @ Wff2 + bff2   (in place on h)
        gemm_kernel<1><<<dim3(768/64, M/256), 256, 0, stream>>>(
            ffb, wff2T, bff2 + (size_t)l * 768, h, h, nullptr,
            nullptr, nullptr, nullptr, M, 768, 3072);
    }

    // final LN + tied logits: logits = LN(h) @ tok_emb^T
    ln_kernel<<<M, 256, 0, stream>>>(h, lnfg, lnfb, hn);
    convert_bf16<<<(256 * 768 + 255) / 256, 256, 0, stream>>>(tok, tokb, 256 * 768);
    // Bt of tok_emb^T is tok_emb itself ([V,K] = [256,768])
    gemm_kernel<0><<<dim3(256/64, M/256), 256, 0, stream>>>(
        hn, tokb, nullptr, nullptr, (float*)d_out, nullptr,
        nullptr, nullptr, nullptr, M, 256, 768);
}
